// PatchTSTEncoder_1047972020495
// MI455X (gfx1250) — matrix operand order repaired and checked by reading, not compiled, not run on hardware
//
#include <hip/hip_runtime.h>
#include <hip/hip_bf16.h>
#include <cstdint>
#include <cstddef>

typedef __attribute__((ext_vector_type(16))) __bf16 v16bf;
typedef __attribute__((ext_vector_type(8)))  float  v8f;

#define TOKENS 4092   // real token count (4 * 1023)
#define TOKPAD 4096   // padded row count: every M-tile is full
#define NPATCH 1023
#define DIMM   512

static __device__ __forceinline__ uint16_t f2bf(float f) {
  uint32_t u = __float_as_uint(f);
  u += 0x7FFFu + ((u >> 16) & 1u);
  return (uint16_t)(u >> 16);
}
static __device__ __forceinline__ float bf2f(uint16_t h) {
  return __uint_as_float(((uint32_t)h) << 16);
}
// tanh-form GELU through the hardware transcendental (v_exp_f32).
static __device__ __forceinline__ float gelu_fast(float v) {
  float z = 1.595769122f * (v + 0.044715f * v * v * v);
  return v / (1.f + __expf(-z));
}

// One fragment = 16 bf16 per lane = two 128-bit loads.
union Frag {
  uint32_t u[8];
  uint4    q[2];
  uint16_t h[16];
  v16bf    v;
};

// --------------------------------------------------------------------------
// Convert f32 weight (K x N) to bf16, tiled into WMMA B-fragment order.
// blockIdx.y = kt;  idx covers one kt-slab of N*32 elements (no div/mod).
__global__ void cvt_tile_w(const float* __restrict__ W, uint16_t* __restrict__ out,
                           int K, int N) {
  int kt  = blockIdx.y;
  int idx = blockIdx.x * blockDim.x + threadIdx.x;
  int slab = N * 32;  // (N/16)*512
  if (idx >= slab) return;
  int j    = idx & 15;
  int lane = (idx >> 4) & 31;
  int nt   = idx >> 9;
  int krow = kt * 32 + (j >> 3) * 16 + (lane >> 4) * 8 + (j & 7);
  int col  = nt * 16 + (lane & 15);
  float v = (krow < K) ? W[(size_t)krow * N + col] : 0.f;
  out[(size_t)kt * slab + idx] = f2bf(v);
}

// ------------------------------------------------------- patch gather + GEMM
__global__ void patch_embed(const float* __restrict__ x,
                            const uint16_t* __restrict__ wembt,  // tiled
                            const float* __restrict__ bemb,
                            float* __restrict__ emb_out,
                            float* __restrict__ h) {
  __shared__ __align__(16) uint16_t At[16][128];
  int m0 = blockIdx.x * 16;
  int tid = threadIdx.x;  // 128 threads = 4 waves
  for (int e = tid; e < 16 * 128; e += 128) {
    int i = e >> 7, j = e & 127;
    float val = 0.f;
    int tok = m0 + i;
    if (j < 112 && tok < TOKENS) {
      int b = tok / NPATCH, p = tok % NPATCH;
      int c = j >> 4, l = j & 15;  // (C, PATCH_LEN) feature layout
      val = x[((size_t)b * 8192 + (size_t)(p * 8 + l)) * 7 + c];
    }
    At[i][j] = f2bf(val);
  }
  __syncthreads();

  int w = tid >> 5, lane = tid & 31;
  int col16 = lane & 15, khalf = lane >> 4;
  int n0 = blockIdx.y * 64 + w * 16;
  const uint16_t* Bp = wembt + ((size_t)(n0 >> 4) * 32 + lane) * 16;
  const size_t bstep = (size_t)DIMM * 32;
  v8f acc = {};
  for (int k0 = 0; k0 < 128; k0 += 32, Bp += bstep) {
    Frag a, b;
    a.q[0] = *(const uint4*)&At[col16][k0 + 8 * khalf];
    a.q[1] = *(const uint4*)&At[col16][k0 + 16 + 8 * khalf];
    b.q[0] = *(const uint4*)(Bp);
    b.q[1] = *(const uint4*)(Bp + 8);
    acc = __builtin_amdgcn_wmma_f32_16x16x32_bf16(false, a.v, false, b.v,
                                                  (short)0, acc, false, false);
  }
  int colg = n0 + col16;
  float bv = bemb[colg];
  size_t o = (size_t)(m0 + 8 * khalf) * DIMM + colg;
  if (m0 + 16 <= TOKENS) {
#pragma unroll
    for (int r = 0; r < 8; ++r, o += DIMM) {
      float vv = acc[r] + bv;
      emb_out[o] = vv;
      h[o] = vv;
    }
  } else {
#pragma unroll
    for (int r = 0; r < 8; ++r, o += DIMM) {
      float vv = acc[r] + bv;
      h[o] = vv;                                   // padded, always safe
      if (m0 + r + 8 * khalf < TOKENS) emb_out[o] = vv;
    }
  }
}

// ------------------------------------------------------------ generic GEMM
// Wave computes 4 N-tiles (64 cols) reusing one A fragment per k-step.
// Block = 8 waves -> 16 x 512 tile. M % 16 == 0; N is compile-time.
template <int RES, int GELU, int OUTF, int OUTBF, int N>
__global__ void gemm_bf16(const uint16_t* __restrict__ A,
                          const uint16_t* __restrict__ Bt,
                          const float* __restrict__ bias,
                          const float* __restrict__ res,
                          float* __restrict__ outf,
                          uint16_t* __restrict__ outbf,
                          int K) {
  int tid = threadIdx.x, w = tid >> 5, lane = tid & 31;
  int col16 = lane & 15, khalf = lane >> 4;
  int m0 = blockIdx.x * 16;
  int n0 = blockIdx.y * 512 + w * 64;
  const uint16_t* Arow = A + (size_t)(m0 + col16) * K;
  const uint16_t* Bp = Bt + ((size_t)(n0 >> 4) * 32 + lane) * 16;
  constexpr size_t bstep = (size_t)N * 32;  // (N/16)*512 elements per kt
  v8f acc[4] = {};
  for (int k0 = 0; k0 < K; k0 += 32, Bp += bstep) {
    Frag a;
    a.q[0] = *(const uint4*)&Arow[k0 + 8 * khalf];
    a.q[1] = *(const uint4*)&Arow[k0 + 16 + 8 * khalf];
    if (k0 + 32 < K) __builtin_prefetch((const void*)(Bp + bstep), 0, 1);
#pragma unroll
    for (int t = 0; t < 4; ++t) {
      Frag b;
      b.q[0] = *(const uint4*)(Bp + t * 512);
      b.q[1] = *(const uint4*)(Bp + t * 512 + 8);
      acc[t] = __builtin_amdgcn_wmma_f32_16x16x32_bf16(
          false, a.v, false, b.v, (short)0, acc[t], false, false);
    }
  }
#pragma unroll
  for (int t = 0; t < 4; ++t) {
    int colg = n0 + t * 16 + col16;
    float bv = bias[colg];
    size_t o = (size_t)(m0 + 8 * khalf) * N + colg;
#pragma unroll
    for (int r = 0; r < 8; ++r) {
      size_t oi = o + (size_t)r * N;        // N constant: immediate offsets
      float vv = acc[t][r] + bv;
      if (RES) vv += res[oi];
      if (GELU) vv = gelu_fast(vv);
      if (OUTF)  outf[oi] = vv;
      if (OUTBF) outbf[oi] = f2bf(vv);
    }
  }
}

// -------------------------------------------------------------- layer norm
__global__ void layernorm_k(const float* __restrict__ h,
                            const float* __restrict__ g,
                            const float* __restrict__ b,
                            uint16_t* __restrict__ obf,
                            float* __restrict__ of, int M) {
  int w = threadIdx.x >> 5, lane = threadIdx.x & 31;
  int tok = blockIdx.x * 8 + w;
  if (tok >= M) return;
  const float4* row4 = (const float4*)(h + (size_t)tok * DIMM) + lane * 4;
  float xv[16];
#pragma unroll
  for (int k2 = 0; k2 < 4; ++k2) {
    float4 t4 = row4[k2];
    xv[k2 * 4 + 0] = t4.x; xv[k2 * 4 + 1] = t4.y;
    xv[k2 * 4 + 2] = t4.z; xv[k2 * 4 + 3] = t4.w;
  }
  float s = 0.f;
#pragma unroll
  for (int j = 0; j < 16; ++j) s += xv[j];
#pragma unroll
  for (int m = 1; m < 32; m <<= 1) s += __shfl_xor(s, m, 32);
  float mu = s * (1.f / 512.f);
  float v = 0.f;
#pragma unroll
  for (int j = 0; j < 16; ++j) { float d = xv[j] - mu; v += d * d; }
#pragma unroll
  for (int m = 1; m < 32; m <<= 1) v += __shfl_xor(v, m, 32);
  float rstd = rsqrtf(v * (1.f / 512.f) + 1e-5f);

  int d0 = lane * 16;
  float y[16];
#pragma unroll
  for (int k2 = 0; k2 < 4; ++k2) {
    float4 g4 = *((const float4*)(g + d0) + k2);
    float4 b4 = *((const float4*)(b + d0) + k2);
    y[k2 * 4 + 0] = (xv[k2 * 4 + 0] - mu) * rstd * g4.x + b4.x;
    y[k2 * 4 + 1] = (xv[k2 * 4 + 1] - mu) * rstd * g4.y + b4.y;
    y[k2 * 4 + 2] = (xv[k2 * 4 + 2] - mu) * rstd * g4.z + b4.z;
    y[k2 * 4 + 3] = (xv[k2 * 4 + 3] - mu) * rstd * g4.w + b4.w;
  }
  if (obf) {
    union { uint16_t hh[16]; uint4 q[2]; } o;
#pragma unroll
    for (int j = 0; j < 16; ++j) o.hh[j] = f2bf(y[j]);
    uint4* dst = (uint4*)(obf + (size_t)tok * DIMM + d0);
    dst[0] = o.q[0]; dst[1] = o.q[1];
  }
  if (of) {
    float4* dst = (float4*)(of + (size_t)tok * DIMM + d0);
#pragma unroll
    for (int k2 = 0; k2 < 4; ++k2) {
      float4 t4;
      t4.x = y[k2 * 4 + 0]; t4.y = y[k2 * 4 + 1];
      t4.z = y[k2 * 4 + 2]; t4.w = y[k2 * 4 + 3];
      dst[k2] = t4;
    }
  }
}

// ------------------------------------------------------------------- RoPE
__global__ void rope_k(uint16_t* __restrict__ qkv) {
  int tid = blockIdx.x * blockDim.x + threadIdx.x;
  if (tid >= TOKENS * 8 * 4) return;
  int jg = (tid & 3) * 8;
  int hh = (tid >> 2) & 7;
  int t  = tid >> 5;
  int pos = t % NPATCH;
  uint16_t* q = qkv + (size_t)t * 1536 + hh * 64;
  uint16_t* k = q + 512;
  union U { uint16_t h[8]; uint4 q4; };
  U q1, q2, k1, k2;
  q1.q4 = *(const uint4*)(q + jg);
  q2.q4 = *(const uint4*)(q + jg + 32);
  k1.q4 = *(const uint4*)(k + jg);
  k2.q4 = *(const uint4*)(k + jg + 32);
#pragma unroll
  for (int c = 0; c < 8; ++c) {
    int j = jg + c;
    float inv = __expf(-(float)j * 0.28782313662425572f);  // ln(1e4)/32
    float sn, cs;
    __sincosf((float)pos * inv, &sn, &cs);
    float a = bf2f(q1.h[c]), b = bf2f(q2.h[c]);
    q1.h[c] = f2bf(a * cs - b * sn);
    q2.h[c] = f2bf(b * cs + a * sn);
    float e = bf2f(k1.h[c]), f = bf2f(k2.h[c]);
    k1.h[c] = f2bf(e * cs - f * sn);
    k2.h[c] = f2bf(f * cs + e * sn);
  }
  *(uint4*)(q + jg)      = q1.q4;
  *(uint4*)(q + jg + 32) = q2.q4;
  *(uint4*)(k + jg)      = k1.q4;
  *(uint4*)(k + jg + 32) = k2.q4;
}

// ------------------------------------------- V transpose: Vt[b][h][d][1024]
__global__ void transpose_v(const uint16_t* __restrict__ qkv,
                            uint16_t* __restrict__ vt) {
  int tid = blockIdx.x * blockDim.x + threadIdx.x;
  if (tid >= 4 * 8 * 64 * 128) return;
  int pg = (tid & 127) * 8;
  int d  = (tid >> 7) & 63;
  int hh = (tid >> 13) & 7;
  int b  = tid >> 16;
  union { uint16_t h[8]; uint4 q; } o;
#pragma unroll
  for (int c = 0; c < 8; ++c) {
    int pos = pg + c;
    o.h[c] = (pos < NPATCH)
                 ? qkv[(size_t)(b * NPATCH + pos) * 1536 + 1024 + hh * 64 + d]
                 : (uint16_t)0;   // zero pad: keeps NaN out of P@V
  }
  *(uint4*)&vt[(((size_t)(b * 8 + hh) * 64 + d) << 10) + pg] = o.q;
}

// ------------------------------------------------- flash attention (WMMA)
__global__ void flash_attn(const uint16_t* __restrict__ qkv,
                           const uint16_t* __restrict__ vt,
                           uint16_t* __restrict__ obf) {
  __shared__ __align__(16) uint16_t pt[4][16][32];
  int tid = threadIdx.x, w = tid >> 5, lane = tid & 31;
  int col16 = lane & 15, khalf = lane >> 4;
  int bh = blockIdx.x; int b = bh >> 3, hh = bh & 7;
  int q0 = (blockIdx.y * 4 + w) * 16;
  const uint16_t* qb  = qkv + (size_t)b * NPATCH * 1536 + hh * 64;
  const uint16_t* vtb = vt + ((size_t)(b * 8 + hh) << 16);  // 64*1024

  // Q fragments (rows >= NPATCH read padded rows; masked at store)
  Frag qf[2];
  {
    const uint16_t* qr = qb + (size_t)(q0 + col16) * 1536;
#pragma unroll
    for (int d = 0; d < 2; ++d) {
      qf[d].q[0] = *(const uint4*)&qr[d * 32 + 16 * khalf];
      qf[d].q[1] = *(const uint4*)&qr[d * 32 + 16 * khalf + 8];
    }
  }

  v8f o0 = {}, o1 = {}, o2 = {}, o3 = {};
  float mrow[8], lrow[8];
#pragma unroll
  for (int r = 0; r < 8; ++r) { mrow[r] = -3.0e38f; lrow[r] = 0.f; }

  for (int kv0 = 0; kv0 < NPATCH; kv0 += 32) {
    v8f s0 = {}, s1 = {};
#pragma unroll
    for (int t = 0; t < 2; ++t) {
      const uint16_t* kr = qb + 512 + (size_t)(kv0 + t * 16 + col16) * 1536;
      v8f sacc = {};
#pragma unroll
      for (int ks = 0; ks < 2; ++ks) {
        Frag bf;
        bf.q[0] = *(const uint4*)&kr[ks * 32 + khalf * 16];
        bf.q[1] = *(const uint4*)&kr[ks * 32 + khalf * 16 + 8];
        sacc = __builtin_amdgcn_wmma_f32_16x16x32_bf16(
            false, qf[ks].v, false, bf.v, (short)0, sacc, false, false);
      }
      if (t == 0) s0 = sacc; else s1 = sacc;
    }

    bool v0ok = (kv0 + col16) < NPATCH;
    bool v1ok = (kv0 + 16 + col16) < NPATCH;
#pragma unroll
    for (int r = 0; r < 8; ++r) {
      float a0 = v0ok ? s0[r] * 0.125f : -3.0e38f;  // scale = 1/sqrt(64)
      float a1 = v1ok ? s1[r] * 0.125f : -3.0e38f;
      float mx = fmaxf(a0, a1);
      mx = fmaxf(mx, __shfl_xor(mx, 1, 32));
      mx = fmaxf(mx, __shfl_xor(mx, 2, 32));
      mx = fmaxf(mx, __shfl_xor(mx, 4, 32));
      mx = fmaxf(mx, __shfl_xor(mx, 8, 32));
      float mnew = fmaxf(mrow[r], mx);
      float corr = __expf(mrow[r] - mnew);
      float p0 = __expf(a0 - mnew);
      float p1 = __expf(a1 - mnew);
      float rs = p0 + p1;
      rs += __shfl_xor(rs, 1, 32);
      rs += __shfl_xor(rs, 2, 32);
      rs += __shfl_xor(rs, 4, 32);
      rs += __shfl_xor(rs, 8, 32);
      lrow[r] = lrow[r] * corr + rs;
      mrow[r] = mnew;
      o0[r] *= corr; o1[r] *= corr; o2[r] *= corr; o3[r] *= corr;
      pt[w][r + 8 * khalf][col16]      = f2bf(p0);
      pt[w][r + 8 * khalf][16 + col16] = f2bf(p1);
    }
    __syncthreads();

    Frag pf;
    pf.q[0] = *(const uint4*)&pt[w][col16][16 * khalf];
    pf.q[1] = *(const uint4*)&pt[w][col16][16 * khalf + 8];

#pragma unroll
    for (int t = 0; t < 4; ++t) {
      const uint16_t* vp =
          vtb + ((size_t)(t * 16 + col16) << 10) + kv0 + khalf * 16;
      Frag vf;
      vf.q[0] = *(const uint4*)vp;
      vf.q[1] = *(const uint4*)(vp + 8);
      v8f& ot = (t == 0 ? o0 : t == 1 ? o1 : t == 2 ? o2 : o3);
      ot = __builtin_amdgcn_wmma_f32_16x16x32_bf16(
          false, pf.v, false, vf.v, (short)0, ot, false, false);
    }
    __syncthreads();
  }

  size_t base = ((size_t)b * NPATCH + q0 + 8 * khalf) * DIMM + hh * 64 + col16;
  if (q0 + 16 <= NPATCH) {
#pragma unroll
    for (int r = 0; r < 8; ++r, base += DIMM) {
      float linv = 1.f / lrow[r];
      obf[base]      = f2bf(o0[r] * linv);
      obf[base + 16] = f2bf(o1[r] * linv);
      obf[base + 32] = f2bf(o2[r] * linv);
      obf[base + 48] = f2bf(o3[r] * linv);
    }
  } else {
#pragma unroll
    for (int r = 0; r < 8; ++r, base += DIMM) {
      if (q0 + r + 8 * khalf < NPATCH) {
        float linv = 1.f / lrow[r];
        obf[base]      = f2bf(o0[r] * linv);
        obf[base + 16] = f2bf(o1[r] * linv);
        obf[base + 32] = f2bf(o2[r] * linv);
        obf[base + 48] = f2bf(o3[r] * linv);
      }
    }
  }
}

// -------------------------------------------------------------------- host
extern "C" void kernel_launch(void* const* d_in, const int* in_sizes, int n_in,
                              void* d_out, int out_size, void* d_ws, size_t ws_size,
                              hipStream_t stream) {
  (void)in_sizes; (void)n_in; (void)out_size; (void)ws_size;
  const float* x      = (const float*)d_in[0];
  const float* W_emb  = (const float*)d_in[1];
  const float* b_emb  = (const float*)d_in[2];
  const float* ln1_g  = (const float*)d_in[3];
  const float* ln1_b  = (const float*)d_in[4];
  const float* W_qkv  = (const float*)d_in[5];
  const float* b_qkv  = (const float*)d_in[6];
  const float* W_proj = (const float*)d_in[7];
  const float* b_proj = (const float*)d_in[8];
  const float* ln2_g  = (const float*)d_in[9];
  const float* ln2_b  = (const float*)d_in[10];
  const float* W_mlp1 = (const float*)d_in[11];
  const float* b_mlp1 = (const float*)d_in[12];
  const float* W_mlp2 = (const float*)d_in[13];
  const float* b_mlp2 = (const float*)d_in[14];
  const float* lnf_g  = (const float*)d_in[15];
  const float* lnf_b  = (const float*)d_in[16];
  float* out = (float*)d_out;

  char* ws = (char*)d_ws;
  size_t off = 0;
  auto alloc = [&](size_t bytes) {
    char* p = ws + off; off += (bytes + 255) & ~(size_t)255; return p;
  };
  uint16_t* wembt = (uint16_t*)alloc((size_t)128 * 512 * 2);
  uint16_t* wqkvt = (uint16_t*)alloc((size_t)4 * 512 * 1536 * 2);
  uint16_t* wprjt = (uint16_t*)alloc((size_t)4 * 512 * 512 * 2);
  uint16_t* wm1t  = (uint16_t*)alloc((size_t)4 * 512 * 2048 * 2);
  uint16_t* wm2t  = (uint16_t*)alloc((size_t)4 * 2048 * 512 * 2);
  float*    h     = (float*)   alloc((size_t)TOKPAD * 512 * 4);
  uint16_t* xln   = (uint16_t*)alloc((size_t)TOKPAD * 512 * 2);
  uint16_t* qkvb  = (uint16_t*)alloc((size_t)TOKPAD * 1536 * 2);
  uint16_t* ob    = (uint16_t*)alloc((size_t)TOKPAD * 512 * 2);
  uint16_t* hid   = (uint16_t*)alloc((size_t)TOKPAD * 2048 * 2);
  uint16_t* vt    = (uint16_t*)alloc((size_t)4 * 8 * 64 * 1024 * 2);

  auto tile = [&](const float* src, uint16_t* dst, int K, int Kpad, int N) {
    cvt_tile_w<<<dim3((N * 32 + 255) / 256, Kpad / 32), 256, 0, stream>>>(
        src, dst, K, N);
  };
  tile(W_emb, wembt, 112, 128, 512);
  for (int i = 0; i < 4; ++i) {
    tile(W_qkv  + (size_t)i * 512 * 1536, wqkvt + (size_t)i * 512 * 1536, 512, 512, 1536);
    tile(W_proj + (size_t)i * 512 * 512,  wprjt + (size_t)i * 512 * 512,  512, 512, 512);
    tile(W_mlp1 + (size_t)i * 512 * 2048, wm1t  + (size_t)i * 512 * 2048, 512, 512, 2048);
    tile(W_mlp2 + (size_t)i * 2048 * 512, wm2t  + (size_t)i * 2048 * 512, 2048, 2048, 512);
  }

  patch_embed<<<dim3(TOKPAD / 16, 8), 128, 0, stream>>>(x, wembt, b_emb, out, h);

  auto ln = [&](const float* src, const float* g, const float* bb,
                uint16_t* obf16, float* of32) {
    layernorm_k<<<dim3(TOKPAD / 8), 256, 0, stream>>>(src, g, bb, obf16, of32,
                                                      TOKENS);
  };

  const dim3 blk(256);
  for (int i = 0; i < 4; ++i) {
    ln(h, ln1_g + i * 512, ln1_b + i * 512, xln, nullptr);
    gemm_bf16<0, 0, 0, 1, 1536><<<dim3(TOKPAD / 16, 3), blk, 0, stream>>>(
        xln, wqkvt + (size_t)i * 512 * 1536, b_qkv + i * 1536,
        nullptr, nullptr, qkvb, 512);
    int rt = TOKENS * 8 * 4;
    rope_k<<<dim3((rt + 255) / 256), blk, 0, stream>>>(qkvb);
    int tv = 4 * 8 * 64 * 128;
    transpose_v<<<dim3((tv + 255) / 256), blk, 0, stream>>>(qkvb, vt);
    flash_attn<<<dim3(32, 16), 128, 0, stream>>>(qkvb, vt, ob);
    gemm_bf16<1, 0, 1, 0, 512><<<dim3(TOKPAD / 16, 1), blk, 0, stream>>>(
        ob, wprjt + (size_t)i * 512 * 512, b_proj + i * 512,
        h, h, nullptr, 512);
    ln(h, ln2_g + i * 512, ln2_b + i * 512, xln, nullptr);
    gemm_bf16<0, 1, 0, 1, 2048><<<dim3(TOKPAD / 16, 4), blk, 0, stream>>>(
        xln, wm1t + (size_t)i * 512 * 2048, b_mlp1 + i * 2048,
        nullptr, nullptr, hid, 512);
    gemm_bf16<1, 0, 1, 0, 512><<<dim3(TOKPAD / 16, 1), blk, 0, stream>>>(
        hid, wm2t + (size_t)i * 2048 * 512, b_mlp2 + i * 512,
        h, h, nullptr, 2048);
  }
  ln(h, lnf_g, lnf_b, nullptr, out + (size_t)TOKENS * 512);
}
